// GraphEdgeAttenNetwork_85392539779072
// MI455X (gfx1250) — matrix operand order repaired and checked by reading, not compiled, not run on hardware
//
#include <hip/hip_runtime.h>
#include <hip/hip_bf16.h>
#include <math.h>

#define NN 20000
#define EE 400000

typedef __attribute__((ext_vector_type(16))) _Float16 v16h;
typedef __attribute__((ext_vector_type(8)))  float    v8f;

__device__ __forceinline__ v8f wmma_f16(v16h a, v16h b, v8f c) {
  // D = A(16x32 f16) * B(32x16 f16) + C(16x16 f32)
  return __builtin_amdgcn_wmma_f32_16x16x32_f16(false, a, false, b, (short)0, c,
                                                false, false);
}

// A-matrix 16x32 f16 K-map per ISA 7.12.2 (lanes 0-15: K 0-7/16-23, lanes 16-31: K 8-15/24-31)
__device__ __forceinline__ int kmapA(int lane, int j) {
  return (lane < 16) ? (j < 8 ? j : j + 8) : (j < 8 ? j + 8 : j + 16);
}

// A fragment from an LDS f16 tile: 16 rows, leading dim lda, starting col k0
// (lowers to 2x ds_load_b128 per lane)
__device__ __forceinline__ v16h load_afrag(const _Float16* S, int lda, int k0, int lane) {
  v16h a; int m = lane & 15;
  #pragma unroll
  for (int j = 0; j < 16; ++j) a[j] = S[m * lda + k0 + kmapA(lane, j)];
  return a;
}

// B fragment from pre-swizzled f16 weights: one contiguous 32B load per lane
// fragment (kt,nt) of a [K,N] weight lives at ((kt*ntiles+nt)*32 + lane)*16
__device__ __forceinline__ v16h load_bfragS(const _Float16* Ws, int ntiles, int kt,
                                            int nt, int lane) {
  return *(const v16h*)(Ws + (((size_t)(kt * ntiles + nt) * 32 + lane) << 4));
}

__device__ __forceinline__ float sigmoidf_(float v) { return 1.f / (1.f + __expf(-v)); }
// monotone uint encoding so atomicMax(uint) == float max
__device__ __forceinline__ unsigned fkey(float f) {
  unsigned u = __float_as_uint(f);
  return (u & 0x80000000u) ? ~u : (u | 0x80000000u);
}
__device__ __forceinline__ float fkey_dec(unsigned k) {
  unsigned u = (k & 0x80000000u) ? (k ^ 0x80000000u) : ~k;
  return __uint_as_float(u);
}

// tiny MLP [in_dim -> 32 -> 8], split across the wave: lane (e,half) computes 16
// hidden units + partial output sums; xor-16 butterfly completes the 8 outputs.
__device__ __forceinline__ void tiny_mlp(const float* u, int in_dim,
                                         const float* w1, const float* b1,
                                         const float* w2, const float* b2,
                                         int half, float out[8]) {
  float part[8];
  #pragma unroll
  for (int k = 0; k < 8; ++k) part[k] = 0.f;
  for (int oo = 0; oo < 16; ++oo) {
    int o = (half << 4) + oo;
    float hv = b1[o];
    #pragma unroll
    for (int i = 0; i < 4; ++i)
      if (i < in_dim) hv += u[i] * w1[i * 32 + o];
    hv = fmaxf(hv, 0.f);
    #pragma unroll
    for (int k = 0; k < 8; ++k) part[k] += hv * w2[o * 8 + k];
  }
  #pragma unroll
  for (int k = 0; k < 8; ++k)
    out[k] = part[k] + __shfl_xor(part[k], 16, 32) + b2[k];
}

// -------------------------------------------- weight swizzle prepass (f32 -> f16 frags)
__global__ void swizzle_w_kernel(const float* __restrict__ W, _Float16* __restrict__ out,
                                 int K, int N) {
  int idx = blockIdx.x * 256 + threadIdx.x;
  if (idx >= K * N) return;
  int j = idx & 15, lane = (idx >> 4) & 31, frag = idx >> 9;
  int ntiles = N >> 4;
  int nt = frag % ntiles, kt = frag / ntiles;
  int k = kt * 32 + ((lane >> 4) << 3) + (j < 8 ? j : j + 8);
  int n = nt * 16 + (lane & 15);
  out[idx] = (_Float16)W[(size_t)k * N + n];
}

// ---------------------------------------------------------------- init agg
__global__ void init_agg_kernel(unsigned* __restrict__ agg) {
  size_t i = (size_t)blockIdx.x * blockDim.x + threadIdx.x;
  if (i < (size_t)NN * 128) agg[i] = 0x007FFFFFu; // fkey(-inf)
}

// ---------------- per-node pass: q/v projections + dual uncertainties, all WMMA,
// ---------------- x tile read once (one wave = 16 nodes)
__global__ __launch_bounds__(128) void node_proj_kernel(
    const float* __restrict__ x,
    const _Float16* __restrict__ pq_ws, const float* pq_b,
    const _Float16* __restrict__ pv_ws, const float* pv_b,
    const _Float16* __restrict__ ua_w1s, const float* ua_b1,
    const float* ua_w2, const float* ua_b2,
    const _Float16* __restrict__ ue_w1s, const float* ue_b1,
    const float* ue_w2, const float* ue_b2,
    _Float16* __restrict__ q_proj, float* __restrict__ v_proj,
    float* __restrict__ node_al, float* __restrict__ node_ep) {
  __shared__ _Float16 xs[4][16 * 128];
  int lane = threadIdx.x & 31, wv = threadIdx.x >> 5;
  int tile = blockIdx.x * 4 + wv;
  if (tile >= NN / 16) return;
  int n0 = tile * 16;
  _Float16* S = xs[wv];
  for (int idx = lane; idx < 16 * 128; idx += 32)
    S[idx] = (_Float16)x[(size_t)(n0 + (idx >> 7)) * 128 + (idx & 127)];
  __builtin_amdgcn_wave_barrier();
  int nn = lane & 15, mb = (lane >> 4) << 3;
  float acc_a[8], acc_e[8];
  #pragma unroll
  for (int r = 0; r < 8; ++r) { acc_a[r] = 0.f; acc_e[r] = 0.f; }
  for (int nt = 0; nt < 8; ++nt) {
    v8f cq = {}; v8f cv = {}; v8f ca = {}; v8f ce = {};
    for (int kc = 0; kc < 4; ++kc) {
      v16h a = load_afrag(S, 128, kc * 32, lane);
      cq = wmma_f16(a, load_bfragS(pq_ws, 8, kc, nt, lane), cq);
      cv = wmma_f16(a, load_bfragS(pv_ws, 8, kc, nt, lane), cv);
      ca = wmma_f16(a, load_bfragS(ua_w1s, 8, kc, nt, lane), ca);
      ce = wmma_f16(a, load_bfragS(ue_w1s, 8, kc, nt, lane), ce);
    }
    float bq = pq_b[nt * 16 + nn], bv = pv_b[nt * 16 + nn];
    float b1a = ua_b1[nt * 16 + nn], b1e = ue_b1[nt * 16 + nn];
    float w2a = ua_w2[nt * 16 + nn], w2e = ue_w2[nt * 16 + nn];
    #pragma unroll
    for (int r = 0; r < 8; ++r) {
      size_t o = (size_t)(n0 + mb + r) * 128 + nt * 16 + nn;
      q_proj[o] = (_Float16)(cq[r] + bq);
      v_proj[o] = cv[r] + bv;
      acc_a[r] += fmaxf(ca[r] + b1a, 0.f) * w2a;
      acc_e[r] += fmaxf(ce[r] + b1e, 0.f) * w2e;
    }
  }
  // reduce over the 16 columns held by the lanes of each half-wave
  #pragma unroll
  for (int r = 0; r < 8; ++r) {
    for (int msk = 1; msk <= 8; msk <<= 1) {
      acc_a[r] += __shfl_xor(acc_a[r], msk, 32);
      acc_e[r] += __shfl_xor(acc_e[r], msk, 32);
    }
  }
  if (nn == 0) {
    #pragma unroll
    for (int r = 0; r < 8; ++r) {
      node_al[n0 + mb + r] = sigmoidf_(acc_a[r] + ua_b2[0]);
      node_ep[n0 + mb + r] = sigmoidf_(acc_e[r] + ue_b2[0]);
    }
  }
}

// ------------------------------------------------- fused per-edge kernel
struct WaveSmem {
  _Float16 ef[16 * 384];   // [x_i | edge_feature | x_j] f16; reused as hatt[16*32]
  _Float16 bufB[16 * 256]; // h1 of edge MLP; reused as q16[16*128] + ep16[16*128]
  float temp[16 * 8];
  float conf[16 * 8];
  int srcI[16];
  int dstI[16];
};

__global__ __launch_bounds__(64) void edge_kernel(
    const float* __restrict__ x, const float* __restrict__ efeat, const int* __restrict__ eidx,
    const _Float16* __restrict__ ne_w1s, const float* ne_b1,
    const _Float16* __restrict__ ne_w2s, const float* ne_b2,
    const _Float16* __restrict__ pe_ws, const float* pe_b,
    const float* up_w1, const float* up_b1, const float* up_w2, const float* up_b2,
    const float* at_w1, const float* at_b1, const float* at_w2, const float* at_b2,
    const float* ug_w1, const float* ug_b1, const float* ug_w2, const float* ug_b2,
    const _Float16* __restrict__ aw1s, const float* ab1,
    const _Float16* __restrict__ aw2s, const float* ab2,
    const float* __restrict__ node_al, const float* __restrict__ node_ep,
    const _Float16* __restrict__ q_proj, const float* __restrict__ v_proj,
    unsigned* __restrict__ agg,
    float* __restrict__ gcn, float* __restrict__ prob_o,
    float* __restrict__ qal_o, float* __restrict__ qep_o,
    float* __restrict__ val_o, float* __restrict__ vep_o,
    float* __restrict__ eunc_o) {
  __shared__ WaveSmem sm[2];
  int lane = threadIdx.x & 31, wv = threadIdx.x >> 5;
  WaveSmem& W = sm[wv];
  int e0 = (blockIdx.x * 2 + wv) * 16;
  int nn = lane & 15, mb = (lane >> 4) << 3;

  // ---- P0: indices + gate MLPs spread over the full wave
  if (lane < 16) {
    int e = e0 + lane;
    W.srcI[lane] = eidx[e];
    W.dstI[lane] = eidx[EE + e];
  }
  __syncthreads();
  {
    int e = lane & 15, half = lane >> 4;
    int s = W.srcI[e], d = W.dstI[e];
    float u[4] = {node_al[s], node_ep[s], node_al[d], node_ep[d]};
    if (half == 0) {
      qal_o[e0 + e] = u[0]; qep_o[e0 + e] = u[1];
      val_o[e0 + e] = u[2]; vep_o[e0 + e] = u[3];
    }
    float o8[8];
    tiny_mlp(u, 4, up_w1, up_b1, up_w2, up_b2, half, o8);
    if (half == 0) {
      #pragma unroll
      for (int k = 0; k < 8; ++k)
        eunc_o[(size_t)(e0 + e) * 8 + k] = sigmoidf_(o8[k]);
    }
    tiny_mlp(u, 4, at_w1, at_b1, at_w2, at_b2, half, o8);
    if (half == 0) {
      #pragma unroll
      for (int k = 0; k < 8; ++k)
        W.temp[e * 8 + k] = __logf(1.f + __expf(o8[k])) + 0.5f; // softplus + 0.5
    }
    tiny_mlp(u, 2, ug_w1, ug_b1, ug_w2, ug_b2, half, o8);
    if (half == 0) {
      #pragma unroll
      for (int k = 0; k < 8; ++k)
        W.conf[e * 8 + k] = sigmoidf_(o8[k]);
    }
  }
  __syncthreads();

  // ---- P1: stage ef = [x_i | edge_feature | x_j] as f16 in LDS
  for (int idx = lane; idx < 16 * 384; idx += 32) {
    int m = idx / 384, c = idx - m * 384;
    float v;
    if (c < 128)      v = x[(size_t)W.srcI[m] * 128 + c];
    else if (c < 256) v = efeat[(size_t)(e0 + m) * 128 + (c - 128)];
    else              v = x[(size_t)W.dstI[m] * 128 + (c - 256)];
    W.ef[idx] = (_Float16)v;
  }
  __syncthreads();

  // ---- P2: h1 = relu(ef @ ne_w1 + b1)   [16,384]x[384,256]
  for (int nt = 0; nt < 16; ++nt) {
    if (nt + 1 < 16) // pull next n-tile's weight frags toward L1/GL2 (global_prefetch_b8)
      __builtin_prefetch(ne_w1s + (((size_t)(nt + 1) * 32 + lane) << 4), 0, 0);
    v8f c = {};
    for (int kc = 0; kc < 12; ++kc)
      c = wmma_f16(load_afrag(W.ef, 384, kc * 32, lane),
                   load_bfragS(ne_w1s, 16, kc, nt, lane), c);
    float b = ne_b1[nt * 16 + nn];
    #pragma unroll
    for (int r = 0; r < 8; ++r)
      W.bufB[(mb + r) * 256 + nt * 16 + nn] = (_Float16)fmaxf(c[r] + b, 0.f);
  }
  __syncthreads();

  // ---- P3: gcn_edge = h1 @ ne_w2 + b2   [16,256]x[256,128]
  for (int nt = 0; nt < 8; ++nt) {
    v8f c = {};
    for (int kc = 0; kc < 8; ++kc)
      c = wmma_f16(load_afrag(W.bufB, 256, kc * 32, lane),
                   load_bfragS(ne_w2s, 8, kc, nt, lane), c);
    float b = ne_b2[nt * 16 + nn];
    #pragma unroll
    for (int r = 0; r < 8; ++r)
      gcn[(size_t)(e0 + mb + r) * 128 + nt * 16 + nn] = c[r] + b;
  }
  __syncthreads();

  // ---- P4: edge projection pe (WMMA) -> ep16; stage q16 from q_proj
  _Float16* q16  = W.bufB;            // h1 is dead now
  _Float16* ep16 = W.bufB + 16 * 128;
  for (int nt = 0; nt < 8; ++nt) {
    v8f c = {};
    for (int kc = 0; kc < 4; ++kc)
      c = wmma_f16(load_afrag(W.ef + 128, 384, kc * 32, lane),
                   load_bfragS(pe_ws, 8, kc, nt, lane), c);
    float b = pe_b[nt * 16 + nn];
    #pragma unroll
    for (int r = 0; r < 8; ++r)
      ep16[(mb + r) * 128 + nt * 16 + nn] = (_Float16)(c[r] + b);
  }
  for (int idx = lane; idx < 16 * 128; idx += 32)
    q16[idx] = q_proj[(size_t)W.srcI[idx >> 7] * 128 + (idx & 127)];
  __syncthreads();

  // ---- P5: attention MLP over (edge,head) rows, K=32 == one WMMA step
  _Float16* hatt = W.ef; // ef dead now
  for (int mt = 0; mt < 8; ++mt) {
    for (int nt = 0; nt < 2; ++nt) {
      v8f c = {};
      v16h a;
      {
        int i = lane & 15, rr = mt * 16 + i, e = rr >> 3, hh = rr & 7;
        #pragma unroll
        for (int j = 0; j < 16; ++j) {
          int ch = kmapA(lane, j);
          a[j] = (ch < 16) ? q16[e * 128 + ch * 8 + hh]
                           : ep16[e * 128 + (ch - 16) * 8 + hh];
        }
      }
      c = wmma_f16(a, load_bfragS(aw1s, 2, 0, nt, lane), c);
      #pragma unroll
      for (int r = 0; r < 8; ++r)
        hatt[(mb + r) * 32 + nt * 16 + nn] =
            (_Float16)fmaxf(c[r] + ab1[nt * 16 + nn], 0.f);
    }
    __syncthreads();
    v8f c2 = {};
    c2 = wmma_f16(load_afrag(hatt, 32, 0, lane), load_bfragS(aw2s, 1, 0, 0, lane), c2);
    #pragma unroll
    for (int r = 0; r < 8; ++r) {
      int m = mb + r, rr = mt * 16 + m, e = rr >> 3, hh = rr & 7;
      float t = W.temp[e * 8 + hh];
      float l = (c2[r] + ab2[nn]) / t;
      float mx = l;
      for (int msk = 1; msk <= 8; msk <<= 1) mx = fmaxf(mx, __shfl_xor(mx, msk, 32));
      float p = __expf(l - mx);
      float sum = p;
      for (int msk = 1; msk <= 8; msk <<= 1) sum += __shfl_xor(sum, msk, 32);
      float prob = p / sum;
      prob_o[(size_t)(e0 + e) * 128 + nn * 8 + hh] = prob;
      int col = nn * 8 + hh;
      float xx = prob * W.conf[e * 8 + hh] *
                 v_proj[(size_t)W.dstI[e] * 128 + col];
      atomicMax(&agg[(size_t)W.srcI[e] * 128 + col], fkey(xx));
    }
    __syncthreads();
  }
}

// ------------------------------------------------- final node MLP (WMMA)
__global__ __launch_bounds__(64) void node_out_kernel(
    const float* __restrict__ x, const unsigned* __restrict__ agg,
    const _Float16* __restrict__ pr_w1s, const float* pr_b1,
    const _Float16* __restrict__ pr_w2s, const float* pr_b2,
    float* __restrict__ outp) {
  __shared__ _Float16 cats[2][16 * 256];
  __shared__ _Float16 hs[2][16 * 256];
  int lane = threadIdx.x & 31, wv = threadIdx.x >> 5;
  int n0 = (blockIdx.x * 2 + wv) * 16;
  _Float16* CS = cats[wv]; _Float16* HS = hs[wv];
  for (int idx = lane; idx < 16 * 256; idx += 32) {
    int m = idx >> 8, c = idx & 255;
    float v;
    if (c < 128) v = x[(size_t)(n0 + m) * 128 + c];
    else {
      float av = fkey_dec(agg[(size_t)(n0 + m) * 128 + (c - 128)]);
      if (!(fabsf(av) < 3.0e38f)) av = 0.f; // empty segments / non-finite -> 0
      v = av;
    }
    CS[idx] = (_Float16)v;
  }
  __syncthreads();
  int nn = lane & 15, mb = (lane >> 4) << 3;
  for (int nt = 0; nt < 16; ++nt) {
    v8f c = {};
    for (int kc = 0; kc < 8; ++kc)
      c = wmma_f16(load_afrag(CS, 256, kc * 32, lane),
                   load_bfragS(pr_w1s, 16, kc, nt, lane), c);
    float b = pr_b1[nt * 16 + nn];
    #pragma unroll
    for (int r = 0; r < 8; ++r)
      HS[(mb + r) * 256 + nt * 16 + nn] = (_Float16)fmaxf(c[r] + b, 0.f);
  }
  __syncthreads();
  for (int nt = 0; nt < 8; ++nt) {
    v8f c = {};
    for (int kc = 0; kc < 8; ++kc)
      c = wmma_f16(load_afrag(HS, 256, kc * 32, lane),
                   load_bfragS(pr_w2s, 8, kc, nt, lane), c);
    float b = pr_b2[nt * 16 + nn];
    #pragma unroll
    for (int r = 0; r < 8; ++r)
      outp[(size_t)(n0 + mb + r) * 128 + nt * 16 + nn] = c[r] + b;
  }
}

// ---------------------------------------------------------------- launch
extern "C" void kernel_launch(void* const* d_in, const int* in_sizes, int n_in,
                              void* d_out, int out_size, void* d_ws, size_t ws_size,
                              hipStream_t stream) {
  (void)in_sizes; (void)n_in; (void)out_size; (void)ws_size;
  const float* x     = (const float*)d_in[0];
  const float* efeat = (const float*)d_in[1];
  const int*   eidx  = (const int*)d_in[2];
  const float* ne_w1 = (const float*)d_in[3];
  const float* ne_b1 = (const float*)d_in[4];
  const float* ne_w2 = (const float*)d_in[5];
  const float* ne_b2 = (const float*)d_in[6];
  const float* ua_w1 = (const float*)d_in[7];
  const float* ua_b1 = (const float*)d_in[8];
  const float* ua_w2 = (const float*)d_in[9];
  const float* ua_b2 = (const float*)d_in[10];
  const float* ue_w1 = (const float*)d_in[11];
  const float* ue_b1 = (const float*)d_in[12];
  const float* ue_w2 = (const float*)d_in[13];
  const float* ue_b2 = (const float*)d_in[14];
  const float* up_w1 = (const float*)d_in[15];
  const float* up_b1 = (const float*)d_in[16];
  const float* up_w2 = (const float*)d_in[17];
  const float* up_b2 = (const float*)d_in[18];
  const float* at_w1 = (const float*)d_in[19];
  const float* at_b1 = (const float*)d_in[20];
  const float* at_w2 = (const float*)d_in[21];
  const float* at_b2 = (const float*)d_in[22];
  const float* aw1   = (const float*)d_in[23];
  const float* ab1   = (const float*)d_in[24];
  const float* aw2   = (const float*)d_in[25];
  const float* ab2   = (const float*)d_in[26];
  const float* pe_w  = (const float*)d_in[27];
  const float* pe_b  = (const float*)d_in[28];
  const float* pq_w  = (const float*)d_in[29];
  const float* pq_b  = (const float*)d_in[30];
  const float* pv_w  = (const float*)d_in[31];
  const float* pv_b  = (const float*)d_in[32];
  const float* ug_w1 = (const float*)d_in[33];
  const float* ug_b1 = (const float*)d_in[34];
  const float* ug_w2 = (const float*)d_in[35];
  const float* ug_b2 = (const float*)d_in[36];
  const float* pr_w1 = (const float*)d_in[37];
  const float* pr_b1 = (const float*)d_in[38];
  const float* pr_w2 = (const float*)d_in[39];
  const float* pr_b2 = (const float*)d_in[40];

  // workspace layout (~26.4 MB)
  char* ws = (char*)d_ws;
  float*    node_al = (float*)(ws);
  float*    node_ep = (float*)(ws + 81920);
  _Float16* q_proj  = (_Float16*)(ws + 163840);
  float*    v_proj  = (float*)(ws + 163840 + 5120000);
  unsigned* agg     = (unsigned*)(ws + 163840 + 5120000 + 10240000);
  char* wsw = ws + 25763840;                      // swizzled f16 weights
  _Float16* ne_w1s = (_Float16*)(wsw + 0);        // 384x256
  _Float16* ne_w2s = (_Float16*)(wsw + 196608);   // 256x128
  _Float16* pe_ws  = (_Float16*)(wsw + 262144);   // 128x128
  _Float16* pq_ws  = (_Float16*)(wsw + 294912);   // 128x128
  _Float16* pv_ws  = (_Float16*)(wsw + 327680);   // 128x128
  _Float16* pr_w1s = (_Float16*)(wsw + 360448);   // 256x256
  _Float16* pr_w2s = (_Float16*)(wsw + 491520);   // 256x128
  _Float16* aw1s   = (_Float16*)(wsw + 557056);   // 32x32
  _Float16* aw2s   = (_Float16*)(wsw + 559104);   // 32x16
  _Float16* ua_w1s = (_Float16*)(wsw + 560128);   // 128x128
  _Float16* ue_w1s = (_Float16*)(wsw + 592896);   // 128x128

  // output layout (flat, return order)
  float* outp   = (float*)d_out;                // [N,128]
  float* gcn    = outp + 2560000;               // [E,128]
  float* prob   = outp + 53760000;              // [E,16,8]
  float* qal    = outp + 104960000;             // [E,1] x4
  float* qep    = outp + 105360000;
  float* valo   = outp + 105760000;
  float* vepo   = outp + 106160000;
  float* eunc   = outp + 106560000;             // [E,8]

  // prepass: swizzle all GEMM weights into WMMA B-fragment layout (f16)
  #define SWZ(W, O, K, N) \
    swizzle_w_kernel<<<((K) * (N) + 255) / 256, 256, 0, stream>>>((W), (O), (K), (N))
  SWZ(ne_w1, ne_w1s, 384, 256);
  SWZ(ne_w2, ne_w2s, 256, 128);
  SWZ(pe_w,  pe_ws,  128, 128);
  SWZ(pq_w,  pq_ws,  128, 128);
  SWZ(pv_w,  pv_ws,  128, 128);
  SWZ(pr_w1, pr_w1s, 256, 256);
  SWZ(pr_w2, pr_w2s, 256, 128);
  SWZ(aw1,   aw1s,   32,  32);
  SWZ(aw2,   aw2s,   32,  16);
  SWZ(ua_w1, ua_w1s, 128, 128);
  SWZ(ue_w1, ue_w1s, 128, 128);
  #undef SWZ

  init_agg_kernel<<<(NN * 128 + 255) / 256, 256, 0, stream>>>(agg);
  node_proj_kernel<<<(NN / 16 + 3) / 4, 128, 0, stream>>>(
      x, pq_ws, pq_b, pv_ws, pv_b,
      ua_w1s, ua_b1, ua_w2, ua_b2, ue_w1s, ue_b1, ue_w2, ue_b2,
      q_proj, v_proj, node_al, node_ep);
  edge_kernel<<<EE / 32, 64, 0, stream>>>(
      x, efeat, eidx,
      ne_w1s, ne_b1, ne_w2s, ne_b2, pe_ws, pe_b,
      up_w1, up_b1, up_w2, up_b2, at_w1, at_b1, at_w2, at_b2,
      ug_w1, ug_b1, ug_w2, ug_b2, aw1s, ab1, aw2s, ab2,
      node_al, node_ep, q_proj, v_proj, agg,
      gcn, prob, qal, qep, valo, vepo, eunc);
  node_out_kernel<<<NN / 32, 64, 0, stream>>>(x, agg, pr_w1s, pr_b1, pr_w2s, pr_b2, outp);
}
